// LSTMModel_42709154791735
// MI455X (gfx1250) — matrix operand order repaired and checked by reading, not compiled, not run on hardware
//
#include <hip/hip_runtime.h>

typedef __attribute__((ext_vector_type(16))) _Float16 v16h;
typedef __attribute__((ext_vector_type(8)))  _Float16 v8h;
typedef __attribute__((ext_vector_type(4)))  _Float16 v4h;
typedef __attribute__((ext_vector_type(8)))  float    v8f;
typedef __attribute__((ext_vector_type(4)))  float    v4f;

// gfx1250 hardware v_tanh_f32 (TRANS32, co-executes with WMMA).
__device__ __forceinline__ float fast_tanhf(float x) {
#if __has_builtin(__builtin_amdgcn_tanhf)
  return __builtin_amdgcn_tanhf(x);
#else
  const float e = __expf(x + x);
  return 1.0f - 2.0f * __builtin_amdgcn_rcpf(e + 1.0f);
#endif
}
__device__ __forceinline__ float fast_sigf(float x) {
  return 0.5f * fast_tanhf(0.5f * x) + 0.5f;   // sigmoid via the tanh unit
}

// Workgroup barrier that drains only DScnt (LDS producer/consumer sync).
// __syncthreads()'s fence also emits s_wait_loadcnt 0x0, which would force
// the cross-step x prefetch (private per-lane register fills, no cross-thread
// visibility needed) to complete at every barrier.  This keeps it in flight.
__device__ __forceinline__ void wg_sync_lds() {
  asm volatile("s_wait_dscnt 0x0\n\t"
               "s_barrier_signal -1\n\t"
               "s_barrier_wait -1" ::: "memory");
}

// ---------------------------------------------------------------------------
// Kernel 1: fused conv1(3->16,k3,SAME) -> maxpool2 -> conv2(16->32,k3,SAME)
//           -> maxpool2.  f32 in, f16 out in [B,512,32] (LSTM feed layout).
// ---------------------------------------------------------------------------
__global__ __launch_bounds__(256)
void conv_front_kernel(const float* __restrict__ x,   // [256,2048,3]
                       const float* __restrict__ w1,  // [16,3,3]
                       const float* __restrict__ b1,  // [16]
                       const float* __restrict__ w2,  // [32,16,3]
                       const float* __restrict__ b2,  // [32]
                       _Float16* __restrict__ out)    // [256,512,32]
{
  __shared__ float xls[518 * 4];
  __shared__ float p1s[258 * 16];
  __shared__ float w1s[16 * 9];
  __shared__ float b1s[16];
  __shared__ float w2s[32 * 48];
  __shared__ float b2s[32];

  const int tid = threadIdx.x;
  const int b   = blockIdx.x;
  const int t4b = blockIdx.y * 128;

  if (tid < 144) w1s[tid] = w1[tid];
  if (tid < 16)  b1s[tid] = b1[tid];
  if (tid < 32)  b2s[tid] = b2[tid];
  for (int i = tid; i < 1536; i += 256) w2s[i] = w2[i];

  const int p0 = 4 * t4b - 3;
  for (int i = tid; i < 518; i += 256) {
    const int gp = p0 + i;
    const bool ok = (gp >= 0) && (gp < 2048);
#pragma unroll
    for (int ch = 0; ch < 3; ++ch)
      xls[i * 4 + ch] = ok ? x[((size_t)b * 2048 + gp) * 3 + ch] : 0.0f;
  }
  __syncthreads();

  for (int it = tid; it < 258 * 16; it += 256) {
    const int ql = it >> 4;
    const int oc = it & 15;
    const int gq = 2 * t4b - 1 + ql;
    float v = 0.0f;
    if (gq >= 0 && gq < 1024) {
      float c0 = b1s[oc], c1 = b1s[oc];
#pragma unroll
      for (int k = 0; k < 3; ++k)
#pragma unroll
        for (int ic = 0; ic < 3; ++ic) {
          const float w = w1s[oc * 9 + ic * 3 + k];
          c0 += xls[(2 * ql + k) * 4 + ic] * w;
          c1 += xls[(2 * ql + 1 + k) * 4 + ic] * w;
        }
      v = fmaxf(c0, c1);
    }
    p1s[ql * 16 + oc] = v;
  }
  __syncthreads();

  for (int it = tid; it < 128 * 32; it += 256) {
    const int oc = it & 31;
    const int tl = it >> 5;
    float c0 = b2s[oc], c1 = b2s[oc];
#pragma unroll
    for (int k = 0; k < 3; ++k)
#pragma unroll
      for (int ic = 0; ic < 16; ++ic) {
        const float w = w2s[oc * 48 + ic * 3 + k];
        c0 += p1s[(2 * tl + k) * 16 + ic] * w;
        c1 += p1s[(2 * tl + 1 + k) * 16 + ic] * w;
      }
    out[((size_t)b * 512 + t4b + tl) * 32 + oc] = (_Float16)fmaxf(c0, c1);
  }
}

// ---------------------------------------------------------------------------
// Kernel 2: LSTM layer.  One block = 16 batch rows, 8 waves; wave w owns gate
// columns [32w, 32w+32).  g = [x_t ; h] @ W^T + b via v_wmma_f32_16x16x32_f16;
// gates through v_tanh_f32.  Weights/biases in registers; h/c in LDS.
// x_{t+1} is prefetched after step t's WMMAs (double buffer, loop unrolled
// x2); the DScnt-only barriers keep that load in flight across the step.
// ---------------------------------------------------------------------------
template <int ISIZE>
__global__ __launch_bounds__(256)
void lstm_layer_kernel(const _Float16* __restrict__ xseq,  // [256,T,ISIZE]
                       const float* __restrict__ w_ih,     // [256,ISIZE]
                       const float* __restrict__ w_hh,     // [256,64]
                       const float* __restrict__ b_ih,     // [256]
                       const float* __restrict__ b_hh,     // [256]
                       _Float16* __restrict__ hseq_out,    // [256,T,64] or null
                       const float* __restrict__ fc_w,     // [5,64] or null
                       const float* __restrict__ fc_b,     // [5] or null
                       float* __restrict__ out,            // [256,5] or null
                       int T)
{
  constexpr int NKX = ISIZE / 32;   // K-tiles from x_t
  constexpr int NK  = NKX + 2;      // + 64 hidden units

  __shared__ _Float16 h_buf[16 * 80];   // row stride 80 halves (16B aligned)
  __shared__ float    c_buf[16 * 64];
  __shared__ float    g_buf[16 * 264];  // row stride 264 floats (16B aligned)

  const int tid  = threadIdx.x;
  const int wid  = tid >> 5;
  const int lane = tid & 31;
  const int lm   = lane & 15;
  const int lhi  = lane >> 4;
  const int khb  = lhi * 8;   // A-layout K base (0 or 8); also D-row base
  const int nb   = wid * 32;
  const int B0   = blockIdx.x * 16;

  // ---- Weight (B operand) fragments + biases, resident in registers.
  // B 32x16 f16 layout: lane col n = lm; lanes 0-15: elems 0-7 = K 0-7, elems 8-15 = K 16-23; lanes 16-31: +8.
  v16h  bfrag[2][NK];
  float bias[2];
#pragma unroll
  for (int nt = 0; nt < 2; ++nt) {
    const int n = nb + nt * 16 + lm;
    bias[nt] = b_ih[n] + b_hh[n];
    const int koff = lhi * 8;
#pragma unroll
    for (int kt = 0; kt < NK; ++kt) {
#pragma unroll
      for (int e = 0; e < 16; ++e) {
        float wv;
        if (kt < NKX) wv = w_ih[(size_t)n * ISIZE + kt * 32 + koff + e + (e & 8)];
        else          wv = w_hh[(size_t)n * 64 + (kt - NKX) * 32 + koff + e + (e & 8)];
        bfrag[nt][kt][e] = (_Float16)wv;
      }
    }
  }

  for (int i = tid; i < 16 * 80; i += 256) h_buf[i] = (_Float16)0.0f;
  for (int i = tid; i < 16 * 64; i += 256) c_buf[i] = 0.0f;
  __syncthreads();

  const int row = (tid * 4) >> 6;   // elementwise ownership: 4 units/thread
  const int un  = (tid * 4) & 63;

  // x A-fragment loader.  16-bit A 16x32 layout: lane row = lm; lanes 0-15:
  // elems 0-7 = K 0-7, elems 8-15 = K 16-23; lanes 16-31: +8.
  auto load_x = [&](int t, v16h (&dst)[NKX]) {
    const _Float16* xr = xseq + ((size_t)(B0 + lm) * T + t) * ISIZE;
#pragma unroll
    for (int kt = 0; kt < NKX; ++kt) {
      const v8h lo = *(const v8h*)(xr + kt * 32 + khb);
      const v8h hi = *(const v8h*)(xr + kt * 32 + khb + 16);
#pragma unroll
      for (int e = 0; e < 8; ++e) { dst[kt][e] = lo[e]; dst[kt][8 + e] = hi[e]; }
    }
  };

  auto step = [&](int t, v16h (&acur)[NKX], v16h (&anext)[NKX]) {
    // h fragments from LDS (same A layout)
    v16h hfrag[2];
    const _Float16* hr = &h_buf[lm * 80];
#pragma unroll
    for (int j = 0; j < 2; ++j) {
      const v8h lo = *(const v8h*)(hr + j * 32 + khb);
      const v8h hi = *(const v8h*)(hr + j * 32 + khb + 16);
#pragma unroll
      for (int e = 0; e < 8; ++e) { hfrag[j][e] = lo[e]; hfrag[j][8 + e] = hi[e]; }
    }

    // g = [x,h] @ W^T + b  (accumulators seeded with bias)
#pragma unroll
    for (int nt = 0; nt < 2; ++nt) {
      v8f acc;
#pragma unroll
      for (int v = 0; v < 8; ++v) acc[v] = bias[nt];
#pragma unroll
      for (int kt = 0; kt < NKX; ++kt)
        acc = __builtin_amdgcn_wmma_f32_16x16x32_f16(false, acur[kt], false, bfrag[nt][kt],
                                                     (short)0, acc, false, false);
#pragma unroll
      for (int j = 0; j < 2; ++j)
        acc = __builtin_amdgcn_wmma_f32_16x16x32_f16(false, hfrag[j], false, bfrag[nt][NKX + j],
                                                     (short)0, acc, false, false);
      // D layout: lane col n = lm; VGPR v -> row khb + v
      const int n = nb + nt * 16 + lm;
#pragma unroll
      for (int v = 0; v < 8; ++v) g_buf[(khb + v) * 264 + n] = acc[v];
    }

    // Prefetch x_{t+1}; stays outstanding across the DScnt-only barriers.
    if (t + 1 < T) load_x(t + 1, anext);

    wg_sync_lds();

    // elementwise gates (PyTorch order i,f,g,o along the 256 dim)
    const v4f gi = *(const v4f*)&g_buf[row * 264 + un];
    const v4f gf = *(const v4f*)&g_buf[row * 264 + 64 + un];
    const v4f gg = *(const v4f*)&g_buf[row * 264 + 128 + un];
    const v4f go = *(const v4f*)&g_buf[row * 264 + 192 + un];
    v4f c = *(const v4f*)&c_buf[row * 64 + un];
    v4h h4;
#pragma unroll
    for (int e = 0; e < 4; ++e) {
      const float i_ = fast_sigf(gi[e]);
      const float f_ = fast_sigf(gf[e]);
      const float g_ = fast_tanhf(gg[e]);
      const float o_ = fast_sigf(go[e]);
      const float cn = f_ * c[e] + i_ * g_;
      c[e]  = cn;
      h4[e] = (_Float16)(o_ * fast_tanhf(cn));
    }
    *(v4f*)&c_buf[row * 64 + un] = c;
    *(v4h*)&h_buf[row * 80 + un] = h4;
    if (hseq_out)
      *(v4h*)(hseq_out + ((size_t)(B0 + row) * T + t) * 64 + un) = h4;

    wg_sync_lds();
  };

  v16h ax0[NKX], ax1[NKX];
  load_x(0, ax0);
  for (int t = 0; t < T; t += 2) {   // T even; ping-pong the x double buffer
    step(t,     ax0, ax1);
    step(t + 1, ax1, ax0);
  }

  // ---- fused FC head on the final hidden state (layer 1 only)
  if (out && tid < 80) {
    const int r = tid / 5;
    const int o = tid - r * 5;
    float acc = fc_b[o];
#pragma unroll
    for (int k = 0; k < 64; ++k) acc += (float)h_buf[r * 80 + k] * fc_w[o * 64 + k];
    out[(size_t)(B0 + r) * 5 + o] = acc;
  }
}

extern "C" void kernel_launch(void* const* d_in, const int* in_sizes, int n_in,
                              void* d_out, int out_size, void* d_ws, size_t ws_size,
                              hipStream_t stream) {
  (void)in_sizes; (void)n_in; (void)out_size; (void)ws_size;
  const float* x       = (const float*)d_in[0];
  const float* conv1_w = (const float*)d_in[1];
  const float* conv1_b = (const float*)d_in[2];
  const float* conv2_w = (const float*)d_in[3];
  const float* conv2_b = (const float*)d_in[4];
  const float* w_ih0   = (const float*)d_in[5];
  const float* w_hh0   = (const float*)d_in[6];
  const float* b_ih0   = (const float*)d_in[7];
  const float* b_hh0   = (const float*)d_in[8];
  const float* w_ih1   = (const float*)d_in[9];
  const float* w_hh1   = (const float*)d_in[10];
  const float* b_ih1   = (const float*)d_in[11];
  const float* b_hh1   = (const float*)d_in[12];
  const float* fc_w    = (const float*)d_in[13];
  const float* fc_b    = (const float*)d_in[14];

  _Float16* ws0 = (_Float16*)d_ws;               // conv output  [256,512,32] f16
  _Float16* ws1 = ws0 + (size_t)256 * 512 * 32;  // layer0 h-seq [256,512,64] f16

  conv_front_kernel<<<dim3(256, 4), 256, 0, stream>>>(x, conv1_w, conv1_b,
                                                      conv2_w, conv2_b, ws0);
  lstm_layer_kernel<32><<<16, 256, 0, stream>>>(ws0, w_ih0, w_hh0, b_ih0, b_hh0,
                                                ws1, nullptr, nullptr, nullptr, 512);
  lstm_layer_kernel<64><<<16, 256, 0, stream>>>(ws1, w_ih1, w_hh1, b_ih1, b_hh1,
                                                nullptr, fc_w, fc_b, (float*)d_out, 512);
}
